// TSDecoder_82875688943907
// MI455X (gfx1250) — matrix operand order repaired and checked by reading, not compiled, not run on hardware
//
#include <hip/hip_runtime.h>

// GRU "TSDecoder": h0 = x@W_lin.T + b_lin; 128 steps of GRU where input == hidden
// after step 0. bf16 WMMA (f32 accumulate) for all GEMMs; f32 carry path.

typedef __attribute__((ext_vector_type(16))) __bf16 v16bf;
typedef __attribute__((ext_vector_type(8)))  float  v8f;

#define B_DIM 512
#define FIN_DIM 1024
#define H_DIM 512
#define TS_DIM 128

union Frag16 {
    v16bf v;
    unsigned short u[16];
    uint4 q[2];
};

static __device__ __forceinline__ unsigned short f2bf(float f) {
    union { float f; unsigned int u; } c;
    c.f = f;
    unsigned int u = c.u;
    // round to nearest even
    u += 0x7FFFu + ((u >> 16) & 1u);
    return (unsigned short)(u >> 16);
}

// A fragment: 16x32 bf16 tile (rows row0..row0+15, K cols kc..kc+31) of row-major [*, ldk].
// ISA 7.12.2: lane m(0-15)/m+16 -> khalf 0/1; VGPR 0-3 = K khalf*8+{0..7} pairs,
// VGPR 4-7 = K khalf*8+16+{0..7} pairs -> two contiguous b128 loads.
static __device__ __forceinline__ v16bf load_a(const unsigned short* __restrict__ base,
                                               int row0, int ldk, int kc, int lane) {
    int m = lane & 15;
    int khalf = lane >> 4;
    const unsigned short* p = base + (size_t)(row0 + m) * ldk + kc + khalf * 8;
    Frag16 f;
    f.q[0] = *(const uint4*)(p);        // K = kc+khalf*8 + 0..7
    f.q[1] = *(const uint4*)(p + 16);   // K = kc+khalf*8 + 16..23
    return f.v;
}

// B fragment: 32x16 bf16 (K x N). B[k][n] = W[row0+n][kc+k] with W row-major [*, ldk].
// Lanes 0-15/16-31 -> khalf 0/1; elements 0-7 = K khalf*8+{0..7}, elements 8-15 = K khalf*8+16+{0..7}
// (same as A) -> two b128 loads.
static __device__ __forceinline__ v16bf load_b(const unsigned short* __restrict__ w,
                                               int row0, int ldk, int kc, int lane) {
    int n = lane & 15;
    int h = lane >> 4;
    const unsigned short* p = w + (size_t)(row0 + n) * ldk + kc + h * 8;
    Frag16 f;
    f.q[0] = *(const uint4*)(p);
    f.q[1] = *(const uint4*)(p + 16);
    return f.v;
}

__global__ void cvt_f32_bf16(const float* __restrict__ src,
                             unsigned short* __restrict__ dst, int n) {
    int i = blockIdx.x * blockDim.x + threadIdx.x;
    if (i < n) dst[i] = f2bf(src[i]);
}

// h0 = x @ W_lin.T + b_lin ; write f32 (carry path) and bf16 (next GEMM input).
__global__ void h0_kernel(const unsigned short* __restrict__ xbf,
                          const unsigned short* __restrict__ wlinbf,
                          const float* __restrict__ b_lin,
                          float* __restrict__ h0f,
                          unsigned short* __restrict__ h0bf) {
    int lane = threadIdx.x;
    int ct = blockIdx.x * 16;   // hidden column tile
    int rt = blockIdx.y * 16;   // batch row tile
    v8f acc = {};
    for (int kc = 0; kc < FIN_DIM; kc += 32) {
        v16bf a = load_a(xbf, rt, FIN_DIM, kc, lane);
        v16bf b = load_b(wlinbf, ct, FIN_DIM, kc, lane);
        acc = __builtin_amdgcn_wmma_f32_16x16x32_bf16(false, a, false, b,
                                                      (short)0, acc, false, false);
    }
    int col = ct + (lane & 15);
    int rbase = rt + (lane >> 4) * 8;
    float bias = b_lin[col];
#pragma unroll
    for (int r = 0; r < 8; ++r) {
        float v = acc[r] + bias;
        int row = rbase + r;
        h0f[(size_t)row * H_DIM + col] = v;
        h0bf[(size_t)row * H_DIM + col] = f2bf(v);
    }
}

// One GRU step. For t>=1 xi == h, so both gi and gh use the same A (hbf_in).
// For t==0 (skip_xi=1) gi is bias-only. h_prev (f32) comes from h0f or d_out[t-1].
__global__ void gru_step_kernel(const unsigned short* __restrict__ hbf_in,
                                const float* __restrict__ hprev_f32, int hprev_ld,
                                const unsigned short* __restrict__ wih,
                                const unsigned short* __restrict__ whh,
                                const float* __restrict__ b_ih,
                                const float* __restrict__ b_hh,
                                float* __restrict__ out_t,
                                unsigned short* __restrict__ hbf_out,
                                int skip_xi) {
    int lane = threadIdx.x;
    int ct = blockIdx.x * 16;
    int rt = blockIdx.y * 16;

    v8f aIR = {}, aIZ = {}, aIN = {};
    v8f aHR = {}, aHZ = {}, aHN = {};

    for (int kc = 0; kc < H_DIM; kc += 32) {
        v16bf a = load_a(hbf_in, rt, H_DIM, kc, lane);
        if (!skip_xi) {
            v16bf b0 = load_b(wih, 0 * H_DIM + ct, H_DIM, kc, lane);
            aIR = __builtin_amdgcn_wmma_f32_16x16x32_bf16(false, a, false, b0, (short)0, aIR, false, false);
            v16bf b1 = load_b(wih, 1 * H_DIM + ct, H_DIM, kc, lane);
            aIZ = __builtin_amdgcn_wmma_f32_16x16x32_bf16(false, a, false, b1, (short)0, aIZ, false, false);
            v16bf b2 = load_b(wih, 2 * H_DIM + ct, H_DIM, kc, lane);
            aIN = __builtin_amdgcn_wmma_f32_16x16x32_bf16(false, a, false, b2, (short)0, aIN, false, false);
        }
        v16bf b3 = load_b(whh, 0 * H_DIM + ct, H_DIM, kc, lane);
        aHR = __builtin_amdgcn_wmma_f32_16x16x32_bf16(false, a, false, b3, (short)0, aHR, false, false);
        v16bf b4 = load_b(whh, 1 * H_DIM + ct, H_DIM, kc, lane);
        aHZ = __builtin_amdgcn_wmma_f32_16x16x32_bf16(false, a, false, b4, (short)0, aHZ, false, false);
        v16bf b5 = load_b(whh, 2 * H_DIM + ct, H_DIM, kc, lane);
        aHN = __builtin_amdgcn_wmma_f32_16x16x32_bf16(false, a, false, b5, (short)0, aHN, false, false);
    }

    int col = ct + (lane & 15);
    int rbase = rt + (lane >> 4) * 8;
    float bihr = b_ih[col];
    float bihz = b_ih[col + H_DIM];
    float bihn = b_ih[col + 2 * H_DIM];
    float bhhr = b_hh[col];
    float bhhz = b_hh[col + H_DIM];
    float bhhn = b_hh[col + 2 * H_DIM];

#pragma unroll
    for (int r = 0; r < 8; ++r) {
        int row = rbase + r;
        float hprev = hprev_f32[(size_t)row * hprev_ld + col];
        float ir = aIR[r] + bihr;
        float hr = aHR[r] + bhhr;
        float rg = 1.0f / (1.0f + __expf(-(ir + hr)));
        float zg = 1.0f / (1.0f + __expf(-(aIZ[r] + bihz + aHZ[r] + bhhz)));
        float hn = aHN[r] + bhhn;
        float ng = tanhf(aIN[r] + bihn + rg * hn);
        float hv = (1.0f - zg) * ng + zg * hprev;
        out_t[(size_t)row * (TS_DIM * H_DIM) + col] = hv;
        hbf_out[(size_t)row * H_DIM + col] = f2bf(hv);
    }
}

extern "C" void kernel_launch(void* const* d_in, const int* in_sizes, int n_in,
                              void* d_out, int out_size, void* d_ws, size_t ws_size,
                              hipStream_t stream) {
    (void)in_sizes; (void)n_in; (void)out_size; (void)ws_size;

    const float* x     = (const float*)d_in[0];   // [512,1024]
    const float* W_lin = (const float*)d_in[1];   // [512,1024]
    const float* b_lin = (const float*)d_in[2];   // [512]
    const float* W_ih  = (const float*)d_in[3];   // [1536,512]
    const float* W_hh  = (const float*)d_in[4];   // [1536,512]
    const float* b_ih  = (const float*)d_in[5];   // [1536]
    const float* b_hh  = (const float*)d_in[6];   // [1536]
    float* out = (float*)d_out;                   // [512, 128*512]

    char* ws = (char*)d_ws;
    size_t off = 0;
    auto alloc = [&](size_t bytes) -> char* {
        char* p = ws + off;
        off = (off + bytes + 255) & ~(size_t)255;
        return p;
    };
    unsigned short* xbf    = (unsigned short*)alloc((size_t)B_DIM * FIN_DIM * 2);
    unsigned short* wlinbf = (unsigned short*)alloc((size_t)H_DIM * FIN_DIM * 2);
    unsigned short* wihbf  = (unsigned short*)alloc((size_t)3 * H_DIM * H_DIM * 2);
    unsigned short* whhbf  = (unsigned short*)alloc((size_t)3 * H_DIM * H_DIM * 2);
    float*          h0f    = (float*)alloc((size_t)B_DIM * H_DIM * 4);
    unsigned short* hbf0   = (unsigned short*)alloc((size_t)B_DIM * H_DIM * 2);
    unsigned short* hbf1   = (unsigned short*)alloc((size_t)B_DIM * H_DIM * 2);

    // f32 -> bf16 conversions (weights resident in L2 afterwards)
    {
        int n;
        n = B_DIM * FIN_DIM;
        cvt_f32_bf16<<<dim3((n + 255) / 256), dim3(256), 0, stream>>>(x, xbf, n);
        n = H_DIM * FIN_DIM;
        cvt_f32_bf16<<<dim3((n + 255) / 256), dim3(256), 0, stream>>>(W_lin, wlinbf, n);
        n = 3 * H_DIM * H_DIM;
        cvt_f32_bf16<<<dim3((n + 255) / 256), dim3(256), 0, stream>>>(W_ih, wihbf, n);
        cvt_f32_bf16<<<dim3((n + 255) / 256), dim3(256), 0, stream>>>(W_hh, whhbf, n);
    }

    // h0 GEMM: 16x16 tiles, one wave (32 threads) per 16x16 tile
    h0_kernel<<<dim3(H_DIM / 16, B_DIM / 16), dim3(32), 0, stream>>>(
        xbf, wlinbf, b_lin, h0f, hbf0);

    // 128 sequential GRU steps; kernel launch is the step barrier.
    for (int t = 0; t < TS_DIM; ++t) {
        const unsigned short* hin = (t & 1) ? hbf1 : hbf0;
        unsigned short* hout      = (t & 1) ? hbf0 : hbf1;
        const float* hprev = (t == 0) ? h0f : (out + (size_t)(t - 1) * H_DIM);
        int hprev_ld       = (t == 0) ? H_DIM : (TS_DIM * H_DIM);
        gru_step_kernel<<<dim3(H_DIM / 16, B_DIM / 16), dim3(32), 0, stream>>>(
            hin, hprev, hprev_ld, wihbf, whhbf, b_ih, b_hh,
            out + (size_t)t * H_DIM, hout, (t == 0) ? 1 : 0);
    }
}
